// Decoder_76708115907037
// MI455X (gfx1250) — compile-verified, matrix operand order checked by reading, not run on hardware
//
#include <hip/hip_runtime.h>
#include <hip/hip_bf16.h>
#include <stdint.h>

typedef __attribute__((ext_vector_type(16))) __bf16 v16bf;
typedef __attribute__((ext_vector_type(8)))  float  v8f;
typedef __attribute__((ext_vector_type(4)))  int    v4i;

// ---------------- problem constants ----------------
static constexpr int BN   = 4;
static constexpr int Hh   = 160;
static constexpr int Ww   = 160;
static constexpr int Cc   = 256;   // trunk channels
static constexpr int CH   = 128;   // half channels
static constexpr int HW   = Hh * Ww;
static constexpr long NPIX = (long)BN * HW;      // 102400 pixels
static constexpr int  NBLK = 256;                // bnstats partial blocks (NPIX/400)

// ---------------- helpers ----------------
__device__ __forceinline__ unsigned short f32_to_bf16(float f) {
  union { float f; unsigned u; } v; v.f = f;
  unsigned r = v.u + 0x7FFFu + ((v.u >> 16) & 1u);   // round-to-nearest-even
  return (unsigned short)(r >> 16);
}
__device__ __forceinline__ float bf16_to_f32(unsigned short h) {
  union { unsigned u; float f; } v; v.u = ((unsigned)h) << 16;
  return v.f;
}
__device__ __forceinline__ v16bf make_v16bf(uint4 lo, uint4 hi) {
  union { unsigned u[8]; v16bf v; } t;
  t.u[0]=lo.x; t.u[1]=lo.y; t.u[2]=lo.z; t.u[3]=lo.w;
  t.u[4]=hi.x; t.u[5]=hi.y; t.u[6]=hi.z; t.u[7]=hi.w;
  return t.v;
}

// ---- CDNA5 async global->LDS copy (guarded; falls back to sync copy) ----
__device__ __forceinline__ void async_copy16(const unsigned* g, unsigned* l) {
#if __has_builtin(__builtin_amdgcn_global_load_async_to_lds_b128)
  __builtin_amdgcn_global_load_async_to_lds_b128(
      (__attribute__((address_space(1))) v4i*)(g),
      (__attribute__((address_space(3))) v4i*)(l), 0, 0);
#else
  *(uint4*)l = *(const uint4*)g;
#endif
}
__device__ __forceinline__ void wait_async() {
#if __has_builtin(__builtin_amdgcn_s_wait_asynccnt)
  __builtin_amdgcn_s_wait_asynccnt(0);
#endif
}

// ---------------- 1) NCHW f32 -> NHWC bf16 ----------------
__global__ void k_to_nhwc_bf16(const float* __restrict__ in, unsigned short* __restrict__ out) {
  const long total = NPIX * Cc;
  for (long i = blockIdx.x * (long)blockDim.x + threadIdx.x; i < total;
       i += (long)gridDim.x * blockDim.x) {
    int  c  = (int)(i & (Cc - 1));
    long p  = i >> 8;                    // pixel index (Cc == 256)
    long nb = p / HW;
    long hw = p - nb * HW;
    out[i] = f32_to_bf16(in[(nb * Cc + c) * (long)HW + hw]);
  }
}

// ---------------- 2) pack OIHW f32 weights into WMMA B-operand tiles ----------------
// tile = (coT, tap, ciT): 16 co x 32 ci, 256 dwords, dword = lane*8 + j
// lane: n = lane&15 (co within tile), khalf = lane>>4
// j:    kbase = (j<4 ? 0 : 16) + khalf*8 + 2*(j&3);  pack(bf16 k, bf16 k+1)
__global__ void k_pack_w(const float* __restrict__ w, unsigned* __restrict__ wp,
                         int Co, int Ci) {
  const int ciTiles = Ci / 32;
  const long total = (long)Co * Ci * 9 / 2;
  long id = blockIdx.x * (long)blockDim.x + threadIdx.x;
  if (id >= total) return;
  int within = (int)(id & 255);
  long tile  = id >> 8;
  int lane = within >> 3, j = within & 7;
  int ciT = (int)(tile % ciTiles);
  long t2 = tile / ciTiles;
  int tap = (int)(t2 % 9);
  int coT = (int)(t2 / 9);
  int n = lane & 15, khalf = lane >> 4;
  int kbase = ((j < 4) ? 0 : 16) + khalf * 8 + 2 * (j & 3);
  int co  = coT * 16 + n;
  int ci0 = ciT * 32 + kbase;
  long base = ((long)co * Ci) * 9;
  unsigned short lo = f32_to_bf16(w[base + (long)ci0       * 9 + tap]);
  unsigned short hi = f32_to_bf16(w[base + (long)(ci0 + 1) * 9 + tap]);
  wp[id] = (unsigned)lo | ((unsigned)hi << 16);
}

// ---------------- 3) implicit-GEMM 3x3 conv via WMMA bf16 ----------------
// 256-thread block = 8 waves. Each wave: 16 pixels x 64 output channels.
// All waves share one co-block of 64 channels; per-tap packed weights (32 KB)
// are double-buffered in LDS via async global->LDS copies. B tiles are
// additionally double-buffered in registers across ciT steps.
template <int CI>
__global__ __launch_bounds__(256)
void k_conv3x3_wmma(const unsigned short* __restrict__ xb,  // [B,H,W,CI] bf16
                    const unsigned*       __restrict__ wp,  // packed tiles
                    float*                __restrict__ out, // [B,H,W,Co] f32
                    int Co) {
  constexpr int ciTiles = CI / 32;               // 8
  constexpr int TILE_DW = 256;                   // dwords per 16x32 tile
  constexpr int TAP_DW  = 4 * ciTiles * TILE_DW; // 8192 dwords = 32 KB per tap
  __shared__ unsigned sbuf[2 * TAP_DW];          // 64 KB double buffer

  const int tid   = threadIdx.x;
  const int wave  = tid >> 5;
  const int lane  = tid & 31;
  const int m     = lane & 15;
  const int khalf = lane >> 4;

  const int seg = blockIdx.x * 8 + wave;      // 6400 row-segments total
  const int x0  = (seg % (Ww / 16)) * 16;
  const int row = seg / (Ww / 16);
  const int y   = row % Hh;
  const int n   = row / Hh;
  const int x   = x0 + m;

  const int coT0 = blockIdx.y * 4;            // 4 co-tiles of 16

  // cooperative async stage of one tap's weights (4 co-chunks of 8 KB)
  auto stage = [&](int t, int parity) {
    const unsigned* src0 = wp + ((size_t)coT0 * 9 + t) * (size_t)(ciTiles * TILE_DW);
#pragma unroll
    for (int k = 0; k < 4; ++k) {
      const unsigned* g = src0 + (size_t)k * 9 * (ciTiles * TILE_DW) + tid * 8;
      unsigned*       l = sbuf + parity * TAP_DW + k * (ciTiles * TILE_DW) + tid * 8;
      async_copy16(g, l);
      async_copy16(g + 4, l + 4);
    }
  };

  v8f acc0 = {}, acc1 = {}, acc2 = {}, acc3 = {};

  stage(0, 0);
  wait_async();
  __syncthreads();

  for (int t = 0; t < 9; ++t) {
    const int parity = t & 1;
    if (t + 1 < 9) stage(t + 1, parity ^ 1);   // prefetch next tap (async)

    const int dy = t / 3 - 1, dx = t % 3 - 1;
    const int yy = y + dy;
    if ((unsigned)yy < (unsigned)Hh) {         // zero row: wave-uniform skip
      const int xx = x + dx;
      const bool ok = (unsigned)xx < (unsigned)Ww;  // per-lane zero pad
      const unsigned short* prow =
          xb + (((size_t)(n * Hh + yy)) * Ww + (ok ? xx : 0)) * CI + khalf * 8;

      // pre-stage all A fragments for this tap (one load clause)
      uint4 alo[ciTiles], ahi[ciTiles];
#pragma unroll
      for (int c = 0; c < ciTiles; ++c) {
        alo[c] = uint4{0, 0, 0, 0};
        ahi[c] = uint4{0, 0, 0, 0};
        if (ok) {
          alo[c] = *(const uint4*)(prow + c * 32);
          ahi[c] = *(const uint4*)(prow + c * 32 + 16);
        }
      }

      const unsigned* lb = sbuf + parity * TAP_DW + lane * 8;

      // register double-buffer for the 4 B tiles of each ciT step
      uint4 blo[2][4], bhi[2][4];
#pragma unroll
      for (int k = 0; k < 4; ++k) {
        const unsigned* b = lb + (size_t)k * ciTiles * TILE_DW;
        blo[0][k] = *(const uint4*)(b);
        bhi[0][k] = *(const uint4*)(b + 4);
      }

#pragma unroll
      for (int c = 0; c < ciTiles; ++c) {
        const int cur = c & 1, nxt = cur ^ 1;
        if (c + 1 < ciTiles) {   // issue next step's B loads before consuming
#pragma unroll
          for (int k = 0; k < 4; ++k) {
            const unsigned* b = lb + (size_t)(c + 1) * TILE_DW + (size_t)k * ciTiles * TILE_DW;
            blo[nxt][k] = *(const uint4*)(b);
            bhi[nxt][k] = *(const uint4*)(b + 4);
          }
        }
        const v16bf A = make_v16bf(alo[c], ahi[c]);
        acc0 = __builtin_amdgcn_wmma_f32_16x16x32_bf16(
            false, A, false, make_v16bf(blo[cur][0], bhi[cur][0]), (short)0, acc0, false, false);
        acc1 = __builtin_amdgcn_wmma_f32_16x16x32_bf16(
            false, A, false, make_v16bf(blo[cur][1], bhi[cur][1]), (short)0, acc1, false, false);
        acc2 = __builtin_amdgcn_wmma_f32_16x16x32_bf16(
            false, A, false, make_v16bf(blo[cur][2], bhi[cur][2]), (short)0, acc2, false, false);
        acc3 = __builtin_amdgcn_wmma_f32_16x16x32_bf16(
            false, A, false, make_v16bf(blo[cur][3], bhi[cur][3]), (short)0, acc3, false, false);
      }
    }

    wait_async();        // staged tap t+1 fully in LDS
    __syncthreads();     // + everyone done reading buffer 'parity'
  }

  // C/D layout: VGPR r -> pixel (r + khalf*8); lane&15 -> co within tile
  float* orow = out + (((size_t)(n * Hh + y)) * Ww + x0) * Co + coT0 * 16 + m;
#pragma unroll
  for (int r = 0; r < 8; ++r) {
    const size_t po = (size_t)(r + khalf * 8) * Co;
    orow[po +  0] = acc0[r];
    orow[po + 16] = acc1[r];
    orow[po + 32] = acc2[r];
    orow[po + 48] = acc3[r];
  }
}

// ---------------- 4a) BN partial sums (coalesced, deterministic) ----------------
// Block b covers pixels [b*400, (b+1)*400); thread c handles channel c.
__global__ void k_bnstats1(const float* __restrict__ act, float* __restrict__ ps,
                           float* __restrict__ pq, int C) {
  const int c = threadIdx.x;         // blockDim.x == C
  const int b = blockIdx.x;
  const long p0 = (long)b * (NPIX / NBLK);
  float s = 0.f, q = 0.f;
  for (long p = p0; p < p0 + (NPIX / NBLK); ++p) {
    float v = act[p * C + c];
    s += v; q += v * v;
  }
  ps[(long)b * C + c] = s;
  pq[(long)b * C + c] = q;
}

// ---------------- 4b) BN finalize: reduce partials, emit scale/shift ----------------
__global__ void k_bnstats2(const float* __restrict__ ps, const float* __restrict__ pq,
                           const float* __restrict__ gamma, const float* __restrict__ beta,
                           float* __restrict__ ss, int C) {
  const int c = blockIdx.x, t = threadIdx.x;   // 256 threads == NBLK
  __shared__ float ls[NBLK], lq[NBLK];
  ls[t] = ps[(long)t * C + c];
  lq[t] = pq[(long)t * C + c];
  __syncthreads();
  for (int off = NBLK / 2; off > 0; off >>= 1) {
    if (t < off) { ls[t] += ls[t + off]; lq[t] += lq[t + off]; }
    __syncthreads();
  }
  if (t == 0) {
    const float inv = 1.0f / (float)NPIX;
    float mean = ls[0] * inv;
    float var  = lq[0] * inv - mean * mean;
    float sc   = gamma[c] * rsqrtf(var + 1e-5f);
    ss[c]      = sc;
    ss[C + c]  = beta[c] - mean * sc;
  }
}

// ---------------- 5) BN apply + ReLU + cvt to bf16 NHWC ----------------
__global__ void k_bn_apply_relu(const float* __restrict__ a, const float* __restrict__ ss,
                                unsigned short* __restrict__ ob, int C, long n4) {
  long i = blockIdx.x * (long)blockDim.x + threadIdx.x;
  if (i >= n4) return;
  const long base = i * 4;
  const int  c    = (int)(base & (C - 1));
  const float4 v  = *(const float4*)(a + base);
  float r0 = fmaxf(v.x * ss[c    ] + ss[C + c    ], 0.f);
  float r1 = fmaxf(v.y * ss[c + 1] + ss[C + c + 1], 0.f);
  float r2 = fmaxf(v.z * ss[c + 2] + ss[C + c + 2], 0.f);
  float r3 = fmaxf(v.w * ss[c + 3] + ss[C + c + 3], 0.f);
  uint2 st;
  st.x = (unsigned)f32_to_bf16(r0) | ((unsigned)f32_to_bf16(r1) << 16);
  st.y = (unsigned)f32_to_bf16(r2) | ((unsigned)f32_to_bf16(r3) << 16);
  *(uint2*)(ob + base) = st;
}

// ---------------- 6) small head convs (direct, weights staged in LDS) ----------------
template <int CO>
__global__ void k_head_conv(const unsigned short* __restrict__ act,  // [B,H,W,CH] bf16
                            const float* __restrict__ w,             // [CO,CH,3,3]
                            const float* __restrict__ bias,
                            float* __restrict__ out) {               // NCHW [B,CO,H,W]
  __shared__ float sw[CO * CH * 9];
  for (int i = threadIdx.x; i < CO * CH * 9; i += blockDim.x) sw[i] = w[i];
  __syncthreads();

  const long pix = blockIdx.x * (long)blockDim.x + threadIdx.x;
  if (pix >= NPIX) return;
  const int n  = (int)(pix / HW);
  const int hw = (int)(pix - (long)n * HW);
  const int y  = hw / Ww, x = hw % Ww;

  float acc[CO];
#pragma unroll
  for (int co = 0; co < CO; ++co) acc[co] = bias[co];

  for (int t = 0; t < 9; ++t) {
    const int yy = y + t / 3 - 1, xx = x + t % 3 - 1;
    if ((unsigned)yy >= (unsigned)Hh || (unsigned)xx >= (unsigned)Ww) continue;
    const unsigned short* p = act + ((size_t)(n * Hh + yy) * Ww + xx) * CH;
    for (int ci = 0; ci < CH; ++ci) {
      const float xv = bf16_to_f32(p[ci]);
#pragma unroll
      for (int co = 0; co < CO; ++co) acc[co] += xv * sw[(co * CH + ci) * 9 + t];
    }
  }
#pragma unroll
  for (int co = 0; co < CO; ++co)
    out[((size_t)(n * CO + co)) * HW + hw] = acc[co];
}

// ---------------- host side ----------------
extern "C" void kernel_launch(void* const* d_in, const int* in_sizes, int n_in,
                              void* d_out, int out_size, void* d_ws, size_t ws_size,
                              hipStream_t stream) {
  (void)in_sizes; (void)n_in; (void)out_size; (void)ws_size;
  const float* fpn = (const float*)d_in[0];
  float* outF = (float*)d_out;

  // workspace bump allocator (256B aligned)
  char* ws = (char*)d_ws;
  size_t off = 0;
  auto alloc = [&](size_t bytes) {
    size_t r = off;
    off = (off + bytes + 255) & ~(size_t)255;
    return r;
  };
  unsigned short* xb  = (unsigned short*)(ws + alloc(NPIX * Cc * 2));  // NHWC bf16 input
  float*          a1f = (float*)(ws + alloc(NPIX * Cc * 4));           // conv1 out f32
  unsigned short* a1b = (unsigned short*)(ws + alloc(NPIX * Cc * 2));  // bn1 bf16
  float*          a2f = (float*)(ws + alloc(NPIX * CH * 4));           // conv2 out f32
  unsigned short* a2b = (unsigned short*)(ws + alloc(NPIX * CH * 2));  // bn2 bf16
  float*          ssb = (float*)(ws + alloc(Cc * 2 * 4));              // scale/shift
  float*          ps  = (float*)(ws + alloc((size_t)NBLK * Cc * 4));   // partial sums
  float*          pq  = (float*)(ws + alloc((size_t)NBLK * Cc * 4));   // partial sumsq
  unsigned*       wpA = (unsigned*)(ws + alloc((size_t)Cc * Cc * 9 * 2)); // packed L1
  unsigned*       wpB = (unsigned*)(ws + alloc((size_t)CH * Cc * 9 * 2)); // packed L2

  // input transpose/convert
  k_to_nhwc_bf16<<<4096, 256, 0, stream>>>(fpn, xb);

  // branch parameter indices into d_in
  const int W1[3] = {1, 11, 19}, G1[3] = {2, 12, 20}, B1[3] = {3, 13, 21};
  const int W2[3] = {4, 14, 22}, G2[3] = {5, 15, 23}, B2[3] = {6, 16, 24};

  const long n4a = NPIX * Cc / 4, n4b = NPIX * CH / 4;
  const int packA_blocks = (int)(((long)Cc * Cc * 9 / 2 + 255) / 256);
  const int packB_blocks = (int)(((long)CH * Cc * 9 / 2 + 255) / 256);

  for (int br = 0; br < 3; ++br) {
    k_pack_w<<<packA_blocks, 256, 0, stream>>>((const float*)d_in[W1[br]], wpA, Cc, Cc);
    k_pack_w<<<packB_blocks, 256, 0, stream>>>((const float*)d_in[W2[br]], wpB, CH, Cc);

    k_conv3x3_wmma<Cc><<<dim3(800, Cc / 64), 256, 0, stream>>>(xb, wpA, a1f, Cc);
    k_bnstats1<<<NBLK, Cc, 0, stream>>>(a1f, ps, pq, Cc);
    k_bnstats2<<<Cc, NBLK, 0, stream>>>(ps, pq, (const float*)d_in[G1[br]],
                                        (const float*)d_in[B1[br]], ssb, Cc);
    k_bn_apply_relu<<<(int)((n4a + 255) / 256), 256, 0, stream>>>(a1f, ssb, a1b, Cc, n4a);

    k_conv3x3_wmma<Cc><<<dim3(800, CH / 64), 256, 0, stream>>>(a1b, wpB, a2f, CH);
    k_bnstats1<<<NBLK, CH, 0, stream>>>(a2f, ps, pq, CH);
    k_bnstats2<<<CH, NBLK, 0, stream>>>(ps, pq, (const float*)d_in[G2[br]],
                                        (const float*)d_in[B2[br]], ssb, CH);
    k_bn_apply_relu<<<(int)((n4b + 255) / 256), 256, 0, stream>>>(a2f, ssb, a2b, CH, n4b);

    if (br == 0) {  // sc branch -> shrink + centroid heads
      k_head_conv<1><<<(int)((NPIX + 255) / 256), 256, 0, stream>>>(
          a2b, (const float*)d_in[7], (const float*)d_in[8], outF + 0);
      k_head_conv<1><<<(int)((NPIX + 255) / 256), 256, 0, stream>>>(
          a2b, (const float*)d_in[9], (const float*)d_in[10], outF + 102400);
    } else if (br == 1) {  // param head
      k_head_conv<2><<<(int)((NPIX + 255) / 256), 256, 0, stream>>>(
          a2b, (const float*)d_in[17], (const float*)d_in[18], outF + 204800);
    } else {  // shift head
      k_head_conv<8><<<(int)((NPIX + 255) / 256), 256, 0, stream>>>(
          a2b, (const float*)d_in[25], (const float*)d_in[26], outF + 409600);
    }
  }
}
